// GAE_53060025975425
// MI455X (gfx1250) — matrix operand order repaired and checked by reading, not compiled, not run on hardware
//
#include <hip/hip_runtime.h>
#include <hip/hip_bf16.h>

// ---------------------------------------------------------------------------
// Problem constants (match reference)
// ---------------------------------------------------------------------------
#define NNODES 50000
#define NEDGES 800000
#define MPAD   50048        // NNODES rounded up to 128 (GEMM tile M padding)
#define FIN    128
#define FHID   128          // HEADS*HID = 4*32, also conv2 out
#define FMLPH  256          // MLP hidden

typedef __attribute__((ext_vector_type(4)))  int    v4i;
typedef __attribute__((ext_vector_type(8)))  float  v8f;
typedef __attribute__((ext_vector_type(8)))  __bf16 v8bf;
typedef __attribute__((ext_vector_type(16))) __bf16 v16bf;

#define AS1 __attribute__((address_space(1)))
#define AS3 __attribute__((address_space(3)))

#if defined(__has_builtin)
#if __has_builtin(__builtin_amdgcn_global_load_async_to_lds_b128)
#define HAVE_ASYNC_LDS 1
#endif
#endif

// ---------------------------------------------------------------------------
// helpers
// ---------------------------------------------------------------------------
__device__ __forceinline__ unsigned short f2bf(float f) {
    unsigned u = __float_as_uint(f);
    unsigned rnd = 0x7FFFu + ((u >> 16) & 1u);   // round-to-nearest-even
    return (unsigned short)((u + rnd) >> 16);
}

// order-preserving float->uint encoding for atomicMax-based segment max
__device__ __forceinline__ unsigned encf(float f) {
    unsigned u = __float_as_uint(f);
    return (u & 0x80000000u) ? ~u : (u | 0x80000000u);
}
__device__ __forceinline__ float decf(unsigned e) {
    unsigned u = (e & 0x80000000u) ? (e & 0x7FFFFFFFu) : ~e;
    return __uint_as_float(u);
}
#define ENC_NEG_INF 0x007FFFFFu   // encf(-inf)

// 16-byte global -> LDS copy: async on CDNA5, sync fallback otherwise
__device__ __forceinline__ void copy_b128_to_lds(const unsigned short* g,
                                                 unsigned short* l) {
#ifdef HAVE_ASYNC_LDS
    __builtin_amdgcn_global_load_async_to_lds_b128((AS1 v4i*)g, (AS3 v4i*)l,
                                                   0, 0);
#else
    *(uint4*)l = *(const uint4*)g;
#endif
}
__device__ __forceinline__ void wait_lds_copies() {
#ifdef HAVE_ASYNC_LDS
    asm volatile("s_wait_asynccnt 0x0" ::: "memory");
#endif
}

// ---------------------------------------------------------------------------
// Tiled bf16 WMMA GEMM:  Out[M x Nout] = A[M x K] * W[K x Nout]
//   A  : bf16, row-major [Mpad x K]    (padded rows readable, never stored)
//   Wt : bf16, row-major [NoutPad x K] (pre-transposed weights)
// Block tile 128x128, 8 waves, each wave -> 16 rows x 128 cols (8 wmma accums)
// Staging: async global->LDS b128 copies (no in-kernel conversion).
// Fused epilogue: optional bias, batchnorm(scale/shift), relu;
//                 optional f32 and/or bf16 outputs.
// ---------------------------------------------------------------------------
#define TM 128
#define TN 128
#define TK 32

__global__ __launch_bounds__(256)
void gat_wmma_gemm(const unsigned short* __restrict__ Abf,
                   const unsigned short* __restrict__ Wt,
                   float* __restrict__ Out, unsigned short* __restrict__ OutBf,
                   int M, int K, int Nout, int ldOut,
                   const float* __restrict__ bias,
                   const float* __restrict__ bn_g, const float* __restrict__ bn_be,
                   const float* __restrict__ bn_m, const float* __restrict__ bn_v,
                   int do_relu)
{
    __shared__ unsigned short As[TM * TK];   // [row][k]
    __shared__ unsigned short Bs[TN * TK];   // [col][k]

    const int tid  = threadIdx.x;
    const int lane = tid & 31;
    const int wave = tid >> 5;
    const int mBase = blockIdx.x * TM;
    const int nBase = blockIdx.y * TN;

    v8f acc[8] = {};

    for (int k0 = 0; k0 < K; k0 += TK) {
        // ---- stage A (128x32 bf16 = 512 b128 chunks) + B (512 chunks) ----
        #pragma unroll
        for (int c = tid; c < 1024; c += 256) {
            if (c < 512) {
                int row = c >> 2, part = c & 3;          // 4 chunks per 64B row
                const unsigned short* g =
                    Abf + (size_t)(mBase + row) * K + k0 + part * 8;
                copy_b128_to_lds(g, &As[row * TK + part * 8]);
            } else {
                int cc = c - 512;
                int row = cc >> 2, part = cc & 3;
                const unsigned short* g =
                    Wt + (size_t)(nBase + row) * K + k0 + part * 8;
                copy_b128_to_lds(g, &Bs[row * TK + part * 8]);
            }
        }
        wait_lds_copies();
        __syncthreads();

        // ---- A fragment (16-bit A 16x32 layout):
        // lanes 0-15: m=lane, elems = K[0..7], K[16..23]
        // lanes16-31: m=lane-16, elems = K[8..15], K[24..31]
        const int m  = lane & 15;
        const int kb = (lane < 16) ? 0 : 8;
        const unsigned short* aptr = &As[(wave * 16 + m) * TK];
        v8bf alo = *(const v8bf*)(aptr + kb);
        v8bf ahi = *(const v8bf*)(aptr + kb + 16);
        v16bf afrag = __builtin_shufflevector(alo, ahi,
            0,1,2,3,4,5,6,7,8,9,10,11,12,13,14,15);

        // ---- B fragments: lane n = lane&15; lanes0-15 K0..7,K16..23, lanes16-31 K8..15,K24..31
        const int kbB = (lane < 16) ? 0 : 8;
        #pragma unroll
        for (int j = 0; j < 8; ++j) {
            const unsigned short* bptr = &Bs[(j * 16 + m) * TK + kbB];
            v8bf blo = *(const v8bf*)(bptr);
            v8bf bhi = *(const v8bf*)(bptr + 16);
            v16bf bfrag = __builtin_shufflevector(blo, bhi,
                0,1,2,3,4,5,6,7,8,9,10,11,12,13,14,15);
            acc[j] = __builtin_amdgcn_wmma_f32_16x16x32_bf16(
                false, afrag, false, bfrag, (short)0, acc[j], false, false);
        }
        __syncthreads();
    }

    // ---- epilogue: C/D layout -> VGPR r: lanes0-15 row r, lanes16-31 row r+8
    const int rsel = (lane < 16) ? 0 : 8;
    const int col0 = lane & 15;
    #pragma unroll
    for (int j = 0; j < 8; ++j) {
        int gc = nBase + j * 16 + col0;
        if (gc >= Nout) continue;
        float b = bias ? bias[gc] : 0.0f;
        float scale = 1.0f, shift = 0.0f;
        if (bn_g) {
            float inv = rsqrtf(bn_v[gc] + 1e-5f);
            scale = bn_g[gc] * inv;
            shift = bn_be[gc] - bn_m[gc] * scale;
        }
        #pragma unroll
        for (int r = 0; r < 8; ++r) {
            int gr = mBase + wave * 16 + rsel + r;
            if (gr >= M) continue;
            float v = acc[j][r] + b;
            if (bn_g) v = v * scale + shift;
            if (do_relu) v = fmaxf(v, 0.0f);
            if (Out)   Out[(size_t)gr * ldOut + gc] = v;
            if (OutBf) OutBf[(size_t)gr * ldOut + gc] = f2bf(v);
        }
    }
}

// ---------------------------------------------------------------------------
// one-time converts
// ---------------------------------------------------------------------------
__global__ void gat_f32_to_bf16(const float* __restrict__ src,
                                unsigned short* __restrict__ dst, int count)
{
    int i = blockIdx.x * blockDim.x + threadIdx.x;
    if (i < count) dst[i] = f2bf(src[i]);
}

// W[K x Nout] f32 -> Wt[Nout x K] bf16 (padded rows left untouched; their
// contributions are masked by the GEMM store guards)
__global__ void gat_w_transpose_bf16(const float* __restrict__ W,
                                     unsigned short* __restrict__ Wt,
                                     int K, int Nout)
{
    int i = blockIdx.x * blockDim.x + threadIdx.x;
    if (i >= K * Nout) return;
    int k = i / Nout, nn = i - k * Nout;
    Wt[(size_t)nn * K + k] = f2bf(W[i]);
}

// ---------------------------------------------------------------------------
// per-node attention dots:  as[n,h] = sum_c h[n,h,c]*a_src[h,c]  (same for ad)
// ---------------------------------------------------------------------------
__global__ void gat_att_node(const float* __restrict__ hfeat,
                             const float* __restrict__ a_src,
                             const float* __restrict__ a_dst,
                             float* __restrict__ as_, float* __restrict__ ad_,
                             int n, int H, int C)
{
    int idx = blockIdx.x * blockDim.x + threadIdx.x;
    if (idx >= n * H) return;
    int node = idx / H, h = idx - node * H;
    const float* hp = hfeat + (size_t)node * (H * C) + h * C;
    float s = 0.0f, d = 0.0f;
    for (int c = 0; c < C; ++c) {
        float v = hp[c];
        s += v * a_src[h * C + c];
        d += v * a_dst[h * C + c];
    }
    as_[idx] = s;
    ad_[idx] = d;
}

// ---------------------------------------------------------------------------
__global__ void gat_fill_u32(unsigned* __restrict__ p, unsigned v, int count)
{
    int i = blockIdx.x * blockDim.x + threadIdx.x;
    if (i < count) p[i] = v;
}

// ---------------------------------------------------------------------------
// per-(edge,head): logit = leakyrelu(as[src]+ad[dst]); segment max via atomicMax
// edges e < E_ come from edge_index; e >= E_ are self loops.
// ---------------------------------------------------------------------------
__global__ void gat_edge_logit_max(const int* __restrict__ ei,
                                   const float* __restrict__ as_,
                                   const float* __restrict__ ad_,
                                   float* __restrict__ logit,
                                   unsigned* __restrict__ mEnc,
                                   int E_, int n, int H)
{
    int idx = blockIdx.x * blockDim.x + threadIdx.x;
    int EE = E_ + n;
    if (idx >= EE * H) return;
    int e = idx / H, h = idx - e * H;
    int src, dst;
    if (e < E_) { src = ei[e]; dst = ei[E_ + e]; }
    else        { src = dst = e - E_; }
    float v = as_[src * H + h] + ad_[dst * H + h];
    v = (v > 0.0f) ? v : 0.2f * v;          // leaky relu
    logit[idx] = v;
    atomicMax(&mEnc[dst * H + h], encf(v));
}

// ---------------------------------------------------------------------------
// per-(edge,head): ex = exp(logit - max[dst]); segment sum via atomicAdd.
// overwrites logit with ex.
// ---------------------------------------------------------------------------
__global__ void gat_edge_expsum(const int* __restrict__ ei,
                                float* __restrict__ logit,
                                const unsigned* __restrict__ mEnc,
                                float* __restrict__ ssum,
                                int E_, int n, int H)
{
    int idx = blockIdx.x * blockDim.x + threadIdx.x;
    int EE = E_ + n;
    if (idx >= EE * H) return;
    int e = idx / H, h = idx - e * H;
    int dst = (e < E_) ? ei[E_ + e] : (e - E_);
    float mx = decf(mEnc[dst * H + h]);
    float ex = __expf(logit[idx] - mx);
    logit[idx] = ex;
    atomicAdd(&ssum[dst * H + h], ex);
}

// ---------------------------------------------------------------------------
// per-(edge,head): alpha = ex / (sum[dst]+eps)   (overwrites ex with alpha)
// ---------------------------------------------------------------------------
__global__ void gat_edge_norm(const int* __restrict__ ei,
                              float* __restrict__ exv,
                              const float* __restrict__ ssum,
                              int E_, int n, int H)
{
    int idx = blockIdx.x * blockDim.x + threadIdx.x;
    int EE = E_ + n;
    if (idx >= EE * H) return;
    int e = idx / H, h = idx - e * H;
    int dst = (e < E_) ? ei[E_ + e] : (e - E_);
    exv[idx] = exv[idx] / (ssum[dst * H + h] + 1e-16f);
}

// ---------------------------------------------------------------------------
// per-(edge,feature) weighted scatter: agg[dst,f] += h[src,f] * alpha[e,h(f)]
// F == 128 for both convs. logC = log2(C): 5 (conv1) / 7 (conv2).
// ---------------------------------------------------------------------------
__global__ void gat_edge_aggregate(const int* __restrict__ ei,
                                   const float* __restrict__ hfeat,
                                   const float* __restrict__ alpha,
                                   float* __restrict__ agg,
                                   int E_, int n, int H, int logC)
{
    int idx = blockIdx.x * blockDim.x + threadIdx.x;
    int EE = E_ + n;
    if (idx >= EE * 128) return;
    int e = idx >> 7;
    int f = idx & 127;
    int h = f >> logC;
    int src, dst;
    if (e < E_) { src = ei[e]; dst = ei[E_ + e]; }
    else        { src = dst = e - E_; }
    float a = alpha[e * H + h];
    atomicAdd(&agg[(size_t)dst * 128 + f], hfeat[(size_t)src * 128 + f] * a);
}

// ---------------------------------------------------------------------------
// elementwise: out = [relu]( bn( in + bias ) ), optional bf16 shadow copy
// ---------------------------------------------------------------------------
__global__ void gat_bias_bn_relu(const float* __restrict__ in,
                                 float* __restrict__ out,
                                 unsigned short* __restrict__ outbf,
                                 const float* __restrict__ b,
                                 const float* __restrict__ g,
                                 const float* __restrict__ be,
                                 const float* __restrict__ m,
                                 const float* __restrict__ v,
                                 int n, int F, int do_relu)
{
    int idx = blockIdx.x * blockDim.x + threadIdx.x;
    if (idx >= n * F) return;
    int f = idx % F;
    float x = in[idx] + (b ? b[f] : 0.0f);
    if (g) {
        float inv = rsqrtf(v[f] + 1e-5f);
        x = (x - m[f]) * (g[f] * inv) + be[f];
    }
    if (do_relu) x = fmaxf(x, 0.0f);
    out[idx] = x;
    if (outbf) outbf[idx] = f2bf(x);
}

// ---------------------------------------------------------------------------
// host-side orchestration
// ---------------------------------------------------------------------------
static inline void launch_gemm(const unsigned short* A, const unsigned short* Wt,
                               float* Out, unsigned short* OutBf,
                               int M, int K, int Nout, int ldOut,
                               const float* bias,
                               const float* g, const float* be,
                               const float* m, const float* v,
                               int relu, hipStream_t s)
{
    dim3 grid((M + TM - 1) / TM, (Nout + TN - 1) / TN);
    gat_wmma_gemm<<<grid, 256, 0, s>>>(A, Wt, Out, OutBf, M, K, Nout, ldOut,
                                       bias, g, be, m, v, relu);
}

extern "C" void kernel_launch(void* const* d_in, const int* in_sizes, int n_in,
                              void* d_out, int out_size, void* d_ws, size_t ws_size,
                              hipStream_t stream)
{
    const int N_ = NNODES, E_ = NEDGES;
    const int EE = E_ + N_;

    const float* x    = (const float*)d_in[0];
    const int*   ei   = (const int*)  d_in[1];
    const float* c1W  = (const float*)d_in[2];
    const float* c1as = (const float*)d_in[3];
    const float* c1ad = (const float*)d_in[4];
    const float* c1b  = (const float*)d_in[5];
    const float* c2W  = (const float*)d_in[6];
    const float* c2as = (const float*)d_in[7];
    const float* c2ad = (const float*)d_in[8];
    const float* c2b  = (const float*)d_in[9];
    const float* bn1g = (const float*)d_in[10];
    const float* bn1be= (const float*)d_in[11];
    const float* bn1m = (const float*)d_in[12];
    const float* bn1v = (const float*)d_in[13];
    // MLPs: per block {W1,b1,W2,b2,bn_g,bn_be,bn_m,bn_v}
    const float* p1[8];  for (int i = 0; i < 8; ++i) p1[i]  = (const float*)d_in[14 + i];
    const float* p2[8];  for (int i = 0; i < 8; ++i) p2[i]  = (const float*)d_in[22 + i];
    const float* p12[8]; for (int i = 0; i < 8; ++i) p12[i] = (const float*)d_in[30 + i];

    float* out = (float*)d_out;

    // ---- workspace carve-up: f32 region, then bf16 region ----
    float* wsf = (float*)d_ws;
    size_t o = 0;
    float* featA = wsf + o; o += (size_t)N_ * 128;   // conv1 h / agg2 / z
    float* featB = wsf + o; o += (size_t)N_ * 128;   // agg1 / h1
    float* featC = wsf + o; o += (size_t)N_ * 128;   // conv2 h
    float*    as1  = wsf + o;              o += (size_t)N_ * 4;
    float*    ad1  = wsf + o;              o += (size_t)N_ * 4;
    unsigned* m1   = (unsigned*)(wsf + o); o += (size_t)N_ * 4;
    float*    s1   = wsf + o;              o += (size_t)N_ * 4;
    float*    elog = wsf + o;              o += (size_t)EE * 4;

    unsigned short* wsu = (unsigned short*)(wsf + o);
    size_t u = 0;
    unsigned short* xbf   = wsu + u; u += (size_t)MPAD * 128;
    unsigned short* h1bf  = wsu + u; u += (size_t)MPAD * 128;
    unsigned short* zbf   = wsu + u; u += (size_t)MPAD * 128;
    unsigned short* hidbf = wsu + u; u += (size_t)MPAD * 256;
    unsigned short* c1Wt  = wsu + u; u += (size_t)128 * 128;
    unsigned short* c2Wt  = wsu + u; u += (size_t)128 * 128;
    unsigned short* pW1t[3]; unsigned short* pW2t[3];
    for (int i = 0; i < 3; ++i) { pW1t[i] = wsu + u; u += (size_t)256 * 128; }
    for (int i = 0; i < 3; ++i) { pW2t[i] = wsu + u; u += (size_t)128 * 256; } // 64->128 pad

    const int TPB = 256;
    #define GRID1(cnt) dim3(((cnt) + TPB - 1) / TPB)

    // ---- one-time precision/layout prep ----
    gat_f32_to_bf16<<<GRID1(N_ * 128), TPB, 0, stream>>>(x, xbf, N_ * 128);
    gat_w_transpose_bf16<<<GRID1(128 * 128), TPB, 0, stream>>>(c1W, c1Wt, 128, 128);
    gat_w_transpose_bf16<<<GRID1(128 * 128), TPB, 0, stream>>>(c2W, c2Wt, 128, 128);
    const float* pw1[3] = { p1[0], p2[0], p12[0] };
    const float* pw2[3] = { p1[2], p2[2], p12[2] };
    const int    n2[3]  = { 64, 64, 128 };
    for (int i = 0; i < 3; ++i) {
        gat_w_transpose_bf16<<<GRID1(128 * 256), TPB, 0, stream>>>(pw1[i], pW1t[i], 128, 256);
        gat_w_transpose_bf16<<<GRID1(256 * n2[i]), TPB, 0, stream>>>(pw2[i], pW2t[i], 256, n2[i]);
    }

    // ===================== conv1: h = x @ W  (N x 128) =====================
    launch_gemm(xbf, c1Wt, featA, nullptr, N_, FIN, FHID, FHID,
                nullptr, nullptr, nullptr, nullptr, nullptr, 0, stream);

    gat_att_node<<<GRID1(N_ * 4), TPB, 0, stream>>>(featA, c1as, c1ad, as1, ad1, N_, 4, 32);
    gat_fill_u32<<<GRID1(N_ * 4), TPB, 0, stream>>>(m1, ENC_NEG_INF, N_ * 4);
    gat_fill_u32<<<GRID1(N_ * 4), TPB, 0, stream>>>((unsigned*)s1, 0u, N_ * 4);
    gat_fill_u32<<<GRID1(N_ * 128), TPB, 0, stream>>>((unsigned*)featB, 0u, N_ * 128);

    gat_edge_logit_max<<<GRID1(EE * 4), TPB, 0, stream>>>(ei, as1, ad1, elog, m1, E_, N_, 4);
    gat_edge_expsum   <<<GRID1(EE * 4), TPB, 0, stream>>>(ei, elog, m1, s1, E_, N_, 4);
    gat_edge_norm     <<<GRID1(EE * 4), TPB, 0, stream>>>(ei, elog, s1, E_, N_, 4);
    gat_edge_aggregate<<<GRID1(EE * 128), TPB, 0, stream>>>(ei, featA, elog, featB,
                                                            E_, N_, 4, 5 /*logC: C=32*/);
    // h1 = relu(bn1(agg + b))  (+ bf16 shadow for next GEMM)
    gat_bias_bn_relu<<<GRID1(N_ * 128), TPB, 0, stream>>>(featB, featB, h1bf, c1b,
                                                          bn1g, bn1be, bn1m, bn1v,
                                                          N_, 128, 1);

    // ===================== conv2: h = h1 @ W (N x 128) =====================
    launch_gemm(h1bf, c2Wt, featC, nullptr, N_, FHID, 128, 128,
                nullptr, nullptr, nullptr, nullptr, nullptr, 0, stream);

    gat_att_node<<<GRID1(N_), TPB, 0, stream>>>(featC, c2as, c2ad, as1, ad1, N_, 1, 128);
    gat_fill_u32<<<GRID1(N_), TPB, 0, stream>>>(m1, ENC_NEG_INF, N_);
    gat_fill_u32<<<GRID1(N_), TPB, 0, stream>>>((unsigned*)s1, 0u, N_);
    gat_fill_u32<<<GRID1(N_ * 128), TPB, 0, stream>>>((unsigned*)featA, 0u, N_ * 128);

    gat_edge_logit_max<<<GRID1(EE), TPB, 0, stream>>>(ei, as1, ad1, elog, m1, E_, N_, 1);
    gat_edge_expsum   <<<GRID1(EE), TPB, 0, stream>>>(ei, elog, m1, s1, E_, N_, 1);
    gat_edge_norm     <<<GRID1(EE), TPB, 0, stream>>>(ei, elog, s1, E_, N_, 1);
    gat_edge_aggregate<<<GRID1(EE * 128), TPB, 0, stream>>>(ei, featC, elog, featA,
                                                            E_, N_, 1, 7 /*logC: C=128*/);
    // z = relu(agg + b2)  (+ bf16 shadow for MLP GEMMs)
    gat_bias_bn_relu<<<GRID1(N_ * 128), TPB, 0, stream>>>(featA, featA, zbf, c2b,
                                                          nullptr, nullptr, nullptr, nullptr,
                                                          N_, 128, 1);

    // ===================== MLP heads =====================
    // out layout: [p1: N*64][p2: N*64][p12: N*128]
    float* outp[3] = { out, out + (size_t)N_ * 64, out + (size_t)N_ * 128 };
    const float* const* pp[3] = { p1, p2, p12 };
    for (int i = 0; i < 3; ++i) {
        // hidden = relu(bn(z@W1 + b1))  -> bf16 only
        launch_gemm(zbf, pW1t[i], nullptr, hidbf, N_, 128, FMLPH, FMLPH,
                    pp[i][1], pp[i][4], pp[i][5], pp[i][6], pp[i][7], 1, stream);
        // out_i = hidden@W2 + b2        -> f32 into d_out
        launch_gemm(hidbf, pW2t[i], outp[i], nullptr, N_, FMLPH, n2[i], n2[i],
                    pp[i][3], nullptr, nullptr, nullptr, nullptr, 0, stream);
    }

    (void)in_sizes; (void)n_in; (void)out_size; (void)ws_size;
}
